// testEdgeDecoder_1717986919017
// MI455X (gfx1250) — matrix operand order repaired and checked by reading, not compiled, not run on hardware
//
#include <hip/hip_runtime.h>
#include <hip/hip_bf16.h>
#include <cstdint>

typedef __attribute__((ext_vector_type(4)))  __bf16 v4bf;
typedef __attribute__((ext_vector_type(8)))  __bf16 v8bf;
typedef __attribute__((ext_vector_type(16))) __bf16 v16bf;
typedef __attribute__((ext_vector_type(8)))  float  v8f;
typedef __attribute__((ext_vector_type(8)))  float  v8fv;

#define HDIM    128
#define TILE_M  16
#define THREADS 256
#define NWAVES  8

// ---------------------------------------------------------------------------
// Pre-pass: fp32 -> bf16 stream conversion of the embedding tables into d_ws.
// 33.3 MB total; halves the per-edge gather bytes of the main kernel.
// ---------------------------------------------------------------------------
__global__ __launch_bounds__(THREADS)
void cvt_f32_to_bf16(const float* __restrict__ in, __bf16* __restrict__ outp, int n8) {
    int i = blockIdx.x * blockDim.x + threadIdx.x;
    const int stride = gridDim.x * blockDim.x;
    for (; i < n8; i += stride) {
        const v8fv f = ((const v8fv*)in)[i];
        v8bf b;
        #pragma unroll
        for (int j = 0; j < 8; ++j) b[j] = (__bf16)f[j];
        ((v8bf*)outp)[i] = b;
    }
}

// butterfly fold: pairs rows (a,b) into lanes selected by `bit`, exchanging
// across xor `mask`; lanes with (lane&bit)==0 end up owning a's row.
__device__ __forceinline__ float fold(float a, float b, int lane, int bit, int mask) {
    const bool hi = (lane & bit) != 0;
    float send = hi ? a : b;
    float keep = hi ? b : a;
    return keep + __shfl_xor(send, mask, 32);
}

// ---------------------------------------------------------------------------
// Main kernel (bf16 embeddings in d_ws): async gather -> LDS, double-buffered,
// 8x v_wmma_f32_16x16x32_bf16 per wave-tile, folded epilogue reduction.
// ---------------------------------------------------------------------------
__global__ __launch_bounds__(THREADS)
void edge_decoder_wmma_async(const __bf16* __restrict__ zg,
                             const __bf16* __restrict__ zd,
                             const long long* __restrict__ row,
                             const long long* __restrict__ col,
                             const float* __restrict__ W1,   // [256,128] fp32
                             const float* __restrict__ b1,
                             const float* __restrict__ W2,
                             const float* __restrict__ b2,
                             float* __restrict__ out,
                             int E, int nTiles)
{
    __shared__ __align__(32) __bf16 sA[2][TILE_M * 2 * HDIM];  // 2 x 8 KB bf16
    __shared__ float sPartial[NWAVES * TILE_M];

    const int tid  = threadIdx.x;
    const int lane = tid & 31;
    const int wave = tid >> 5;
    const int half = lane >> 4;
    const int l15  = lane & 15;
    const int n_g  = wave * 16 + l15;

    const float bias = b1[n_g];
    const float w2v  = W2[n_g];
    const float b2v  = b2[0];

    // ---- B fragments (this wave's 16 W1 columns) live in VGPRs all kernel ----
    v16bf Bfrag[8];
    #pragma unroll
    for (int ks = 0; ks < 8; ++ks) {
        const int kbase = ks * 32 + half * 16;
        #pragma unroll
        for (int j = 0; j < 16; ++j)
            Bfrag[ks][j] = (__bf16)W1[(kbase + j) * HDIM + n_g];
    }

    // staging roles: 16 threads per edge, 32 B (2 x b128) per thread
    const int  m      = tid >> 4;
    const int  q      = tid & 15;
    const bool isGene = (q < 8);
    const int  fbase  = (isGene ? 0 : HDIM) + (q & 7) * 16;   // bf16 elements

    auto issue_gather = [&](int tile, int buf) {
        int e = tile * TILE_M + m;
        if (e >= E) e = E - 1;
        const long long idx = isGene ? row[e] : col[e];
        const __bf16* gp = (isGene ? zg : zd) + idx * (long long)HDIM + (q & 7) * 16;
        const unsigned lds0 = (unsigned)(unsigned long long)(&sA[buf][m * (2 * HDIM) + fbase]);
        // CDNA5 async DMA: memory -> LDS, no VGPR round trip, tracked by ASYNCcnt
        asm volatile("global_load_async_to_lds_b128 %0, %1, off"
                     :: "v"(lds0), "v"(gp) : "memory");
        asm volatile("global_load_async_to_lds_b128 %0, %1, off offset:16"
                     :: "v"(lds0 + 16u), "v"(gp) : "memory");
    };

    const int tile0 = blockIdx.x;
    if (tile0 < nTiles) issue_gather(tile0, 0);

    int buf = 0;
    for (int tile = tile0; tile < nTiles; tile += gridDim.x) {
        // current buffer's async DMA done (this wave), then all waves' DMA done
        asm volatile("s_wait_asynccnt 0x0" ::: "memory");
        __syncthreads();

        const int ntile = tile + gridDim.x;
        if (ntile < nTiles) issue_gather(ntile, buf ^ 1);   // overlaps with compute

        // ---- C(16x16) = sum_ks A(16x32) x B(32x16), fp32 accum ----
        v8f c = {};
        const __bf16* arow = &sA[buf][l15 * (2 * HDIM)];
        union AF { v16bf v; v8bf h[2]; } af[8];
        #pragma unroll
        for (int ks = 0; ks < 8; ++ks) {
            af[ks].h[0] = *(const v8bf*)(arow + ks * 32 + half * 16);
            af[ks].h[1] = *(const v8bf*)(arow + ks * 32 + half * 16 + 8);
        }
        #pragma unroll
        for (int ks = 0; ks < 8; ++ks)
            c = __builtin_amdgcn_wmma_f32_16x16x32_bf16(
                    false, af[ks].v, false, Bfrag[ks], (short)0, c, false, false);

        // ---- epilogue: bias + ReLU + *W2[n]; fold 8 rows x 16 lanes -> 8 lanes ----
        float t[8];
        #pragma unroll
        for (int r = 0; r < 8; ++r) t[r] = fmaxf(c[r] + bias, 0.f) * w2v;
        float s0 = fold(t[0], t[1], lane, 1, 1);
        float s1 = fold(t[2], t[3], lane, 1, 1);
        float s2 = fold(t[4], t[5], lane, 1, 1);
        float s3 = fold(t[6], t[7], lane, 1, 1);
        float u0 = fold(s0, s1, lane, 2, 2);
        float u1 = fold(s2, s3, lane, 2, 2);
        float v0 = fold(u0, u1, lane, 4, 4);
        v0 += __shfl_xor(v0, 8, 32);
        // lane now holds the sum for row (l15&7) + 8*half over this wave's 16 cols
        if (l15 < 8) sPartial[wave * TILE_M + half * 8 + l15] = v0;
        __syncthreads();

        if (tid < TILE_M) {
            float s = b2v;
            #pragma unroll
            for (int w = 0; w < NWAVES; ++w) s += sPartial[w * TILE_M + tid];
            const int eo = tile * TILE_M + tid;
            if (eo < E) out[eo] = s;
        }
        buf ^= 1;
    }
}

// ---------------------------------------------------------------------------
// Fallback (ws too small): fp32 gather with in-flight bf16 conversion (round-1)
// ---------------------------------------------------------------------------
__global__ __launch_bounds__(THREADS)
void edge_decoder_wmma_sync(const float* __restrict__ zg,
                            const float* __restrict__ zd,
                            const long long* __restrict__ row,
                            const long long* __restrict__ col,
                            const float* __restrict__ W1,
                            const float* __restrict__ b1,
                            const float* __restrict__ W2,
                            const float* __restrict__ b2,
                            float* __restrict__ out,
                            int E, int nTiles)
{
    __shared__ __align__(32) __bf16 sA[TILE_M * 2 * HDIM];
    __shared__ float sPartial[NWAVES * TILE_M];

    const int tid  = threadIdx.x;
    const int lane = tid & 31;
    const int wave = tid >> 5;
    const int half = lane >> 4;
    const int l15  = lane & 15;
    const int n_g  = wave * 16 + l15;

    const float bias = b1[n_g];
    const float w2v  = W2[n_g];
    const float b2v  = b2[0];

    v16bf Bfrag[8];
    #pragma unroll
    for (int ks = 0; ks < 8; ++ks) {
        const int kbase = ks * 32 + half * 16;
        #pragma unroll
        for (int j = 0; j < 16; ++j)
            Bfrag[ks][j] = (__bf16)W1[(kbase + j) * HDIM + n_g];
    }

    const int  m      = tid >> 4;
    const int  q      = tid & 15;
    const bool isGene = (q < 8);
    const int  fbase  = (isGene ? 0 : HDIM) + (q & 7) * 16;

    for (int tile = blockIdx.x; tile < nTiles; tile += gridDim.x) {
        int e = tile * TILE_M + m;
        if (e >= E) e = E - 1;
        const long long idx = isGene ? row[e] : col[e];
        const float4* src4 =
            (const float4*)((isGene ? zg : zd) + idx * (long long)HDIM) + (q & 7) * 4;
        __bf16* dst = &sA[m * (2 * HDIM) + fbase];
        #pragma unroll
        for (int i = 0; i < 4; ++i) {
            const float4 v = src4[i];
            v4bf b;
            b[0] = (__bf16)v.x; b[1] = (__bf16)v.y;
            b[2] = (__bf16)v.z; b[3] = (__bf16)v.w;
            *(v4bf*)(dst + i * 4) = b;
        }
        __syncthreads();

        v8f c = {};
        const __bf16* arow = &sA[l15 * (2 * HDIM)];
        #pragma unroll
        for (int ks = 0; ks < 8; ++ks) {
            union AF { v16bf v; v8bf h[2]; } af;
            af.h[0] = *(const v8bf*)(arow + ks * 32 + half * 16);
            af.h[1] = *(const v8bf*)(arow + ks * 32 + half * 16 + 8);
            c = __builtin_amdgcn_wmma_f32_16x16x32_bf16(
                    false, af.v, false, Bfrag[ks], (short)0, c, false, false);
        }

        float t[8];
        #pragma unroll
        for (int r = 0; r < 8; ++r) t[r] = fmaxf(c[r] + bias, 0.f) * w2v;
        float s0 = fold(t[0], t[1], lane, 1, 1);
        float s1 = fold(t[2], t[3], lane, 1, 1);
        float s2 = fold(t[4], t[5], lane, 1, 1);
        float s3 = fold(t[6], t[7], lane, 1, 1);
        float u0 = fold(s0, s1, lane, 2, 2);
        float u1 = fold(s2, s3, lane, 2, 2);
        float v0 = fold(u0, u1, lane, 4, 4);
        v0 += __shfl_xor(v0, 8, 32);
        if (l15 < 8) sPartial[wave * TILE_M + half * 8 + l15] = v0;
        __syncthreads();

        if (tid < TILE_M) {
            float s = b2v;
            #pragma unroll
            for (int w = 0; w < NWAVES; ++w) s += sPartial[w * TILE_M + tid];
            const int eo = tile * TILE_M + tid;
            if (eo < E) out[eo] = s;
        }
    }
}

extern "C" void kernel_launch(void* const* d_in, const int* in_sizes, int n_in,
                              void* d_out, int out_size, void* d_ws, size_t ws_size,
                              hipStream_t stream) {
    const float*     zg  = (const float*)d_in[0];
    const float*     zd  = (const float*)d_in[1];
    const long long* row = (const long long*)d_in[2];
    const long long* col = (const long long*)d_in[3];
    const float*     W1  = (const float*)d_in[4];
    const float*     b1  = (const float*)d_in[5];
    const float*     W2  = (const float*)d_in[6];
    const float*     b2  = (const float*)d_in[7];
    float* out = (float*)d_out;

    const int E      = in_sizes[2];
    const int nTiles = (E + TILE_M - 1) / TILE_M;
    const int blocks = nTiles < 2048 ? nTiles : 2048;

    const size_t ngene = (size_t)in_sizes[0];   // N_GENE * H floats
    const size_t ndis  = (size_t)in_sizes[1];   // N_DIS  * H floats
    const size_t need  = (ngene + ndis) * sizeof(__bf16);

    if (ws_size >= need) {
        __bf16* zg_bf = (__bf16*)d_ws;
        __bf16* zd_bf = zg_bf + ngene;
        cvt_f32_to_bf16<<<1024, THREADS, 0, stream>>>(zg, zg_bf, (int)(ngene / 8));
        cvt_f32_to_bf16<<<1024, THREADS, 0, stream>>>(zd, zd_bf, (int)(ndis / 8));
        edge_decoder_wmma_async<<<blocks, THREADS, 0, stream>>>(
            zg_bf, zd_bf, row, col, W1, b1, W2, b2, out, E, nTiles);
    } else {
        edge_decoder_wmma_sync<<<blocks, THREADS, 0, stream>>>(
            zg, zd, row, col, W1, b1, W2, b2, out, E, nTiles);
    }
}
